// Model_33990371181080
// MI455X (gfx1250) — matrix operand order repaired and checked by reading, not compiled, not run on hardware
//
#include <hip/hip_runtime.h>
#include <hip/hip_bf16.h>
#include <stdint.h>

// ---------------------------------------------------------------------------
// BiLSTM "last output" kernel for MI455X (gfx1250).
//   out[:, 0:512]   = relu(forward LSTM final hidden state)   -> full T=512 recurrence
//   out[:, 512:1024]= relu(one backward LSTM step on x[:,T-1]) -> single step
// Forward recurrence: persistent kernel, bf16 WMMA (v_wmma_f32_16x16x32_bf16),
// f32 accumulate, device-wide barrier between GEMM and cell phases.
// ---------------------------------------------------------------------------

typedef __attribute__((ext_vector_type(16))) __bf16         v16bf;
typedef __attribute__((ext_vector_type(8)))  float          v8f;
typedef __attribute__((ext_vector_type(8)))  unsigned short v8us;
typedef __attribute__((ext_vector_type(16))) unsigned short v16us;

static constexpr int BB = 32;     // batch
static constexpr int TT = 512;    // time steps
static constexpr int II = 768;    // input size
static constexpr int HH = 512;    // hidden
static constexpr int GG = 4 * HH; // 2048 gate columns

// f32 -> bf16, round-to-nearest-even
__device__ __forceinline__ unsigned short f2bf(float f) {
  union { float f; unsigned u; } v; v.f = f;
  unsigned u = v.u;
  unsigned r = u + 0x7FFFu + ((u >> 16) & 1u);
  return (unsigned short)(r >> 16);
}

__device__ __forceinline__ float sigmoidf(float x) {
  return 1.0f / (1.0f + __expf(-x));
}

// A fragment (16-bit A 16x32, ISA 7.12.2): lane L holds row m0+(L%16);
// lanes 0-15: K = k0+{0..7, 16..23}; lanes 16-31: K = k0+{8..15, 24..31}.
__device__ __forceinline__ v16bf load_a_frag(const unsigned short* __restrict__ row,
                                             int k0, int lane) {
  const int ko = (lane & 16) ? 8 : 0;
  v8us lo = *reinterpret_cast<const v8us*>(row + k0 + ko);
  v8us hi = *reinterpret_cast<const v8us*>(row + k0 + 16 + ko);
  v16us u;
#pragma unroll
  for (int j = 0; j < 8; ++j) { u[j] = lo[j]; u[j + 8] = hi[j]; }
  union { v16us u; v16bf b; } cvt; cvt.u = u;
  return cvt.b;
}

// B fragment (16-bit B 32x16): lane L holds column n0+(L%16);
// lanes 0-15: K = k0+{0..7, 16..23}; lanes 16-31: K = k0+{8..15, 24..31}.
// B[k][n] = W[n][k], so a column of B is a contiguous row of W.
__device__ __forceinline__ v16bf load_b_frag(const unsigned short* __restrict__ wrow,
                                             int k0, int lane) {
  const int ko = (lane & 16) ? 8 : 0;
  v8us lo = *reinterpret_cast<const v8us*>(wrow + k0 + ko);
  v8us hi = *reinterpret_cast<const v8us*>(wrow + k0 + 16 + ko);
  v16us u;
#pragma unroll
  for (int j = 0; j < 8; ++j) { u[j] = lo[j]; u[j + 8] = hi[j]; }
  union { v16us u; v16bf b; } cvt; cvt.u = u;
  return cvt.b;
}

// Device-wide barrier over gridDim.x blocks (monotone counter, reset by prep).
__device__ __forceinline__ void grid_barrier(unsigned* __restrict__ bar,
                                             unsigned* gen) {
  __threadfence();
  __syncthreads();
  if (threadIdx.x == 0) {
    const unsigned target = (++(*gen)) * gridDim.x;
    __hip_atomic_fetch_add(bar, 1u, __ATOMIC_ACQ_REL, __HIP_MEMORY_SCOPE_AGENT);
    while (__hip_atomic_load(bar, __ATOMIC_ACQUIRE, __HIP_MEMORY_SCOPE_AGENT) < target) {
      __builtin_amdgcn_s_sleep(1);
    }
  }
  __syncthreads();
  __threadfence();  // invalidate stale near caches before reading peer data
}

// ---------------------------------------------------------------------------
// Kernel 1: f32 -> bf16 conversions + barrier reset.
// ---------------------------------------------------------------------------
__global__ void prep_kernel(const float* __restrict__ x,
                            const float* __restrict__ h0,
                            const float* __restrict__ wihf,
                            const float* __restrict__ whhf,
                            unsigned short* __restrict__ xb,
                            unsigned short* __restrict__ wih,
                            unsigned short* __restrict__ whh,
                            unsigned short* __restrict__ hb,
                            unsigned* __restrict__ bar) {
  const long long i = (long long)blockIdx.x * blockDim.x + threadIdx.x;
  if (i == 0) bar[0] = 0u;
  const long long NX = (long long)BB * TT * II;
  const long long NW1 = (long long)GG * II;
  const long long NW2 = (long long)GG * HH;
  const long long NH = (long long)BB * HH;
  if (i < NX) {
    xb[i] = f2bf(x[i]);
  } else if (i < NX + NW1) {
    wih[i - NX] = f2bf(wihf[i - NX]);
  } else if (i < NX + NW1 + NW2) {
    whh[i - NX - NW1] = f2bf(whhf[i - NX - NW1]);
  } else if (i < NX + NW1 + NW2 + NH) {
    hb[i - NX - NW1 - NW2] = f2bf(h0[i - NX - NW1 - NW2]);  // dir 0
  }
}

// ---------------------------------------------------------------------------
// Kernel 2: persistent forward LSTM. 32 blocks x 256 threads = 256 waves.
// Each wave owns one 16x16 tile of gates[32][2048]; per step:
//   gates = xb_t @ Wih^T + h @ Whh^T      (40 WMMAs per tile)
//   barrier; elementwise cell (c lives in registers); barrier.
// ---------------------------------------------------------------------------
__global__ void __launch_bounds__(256)
lstm_fwd_persistent(const unsigned short* __restrict__ xb,
                    const unsigned short* __restrict__ wih,
                    const unsigned short* __restrict__ whh,
                    unsigned short* __restrict__ hb,
                    float* __restrict__ gates,
                    unsigned* __restrict__ bar,
                    const float* __restrict__ c0,   // dir-0 base, [B*H]
                    const float* __restrict__ bih,
                    const float* __restrict__ bhh,
                    float* __restrict__ out) {
  const int lane = threadIdx.x & 31;
  const int wib  = threadIdx.x >> 5;
  const int w    = blockIdx.x * 8 + wib;      // 0..255 tiles
  const int m0   = (w & 1) * 16;              // 2 row tiles (B=32)
  const int n0   = (w >> 1) * 16;             // 128 col tiles (4H=2048)
  const int l15  = lane & 15;
  const int rowB = m0 + l15;
  const unsigned short* wihRow = wih + (size_t)(n0 + l15) * II;
  const unsigned short* whhRow = whh + (size_t)(n0 + l15) * HH;
  const unsigned short* xrowBase = xb + (size_t)rowB * TT * II;

  // Cell ownership: thread e handles elements gtid and gtid+8192 of [B*H]=16384.
  const int gtid = blockIdx.x * 256 + threadIdx.x;
  float creg[2];
#pragma unroll
  for (int e = 0; e < 2; ++e) creg[e] = c0[gtid + e * 8192];

  unsigned gen = 0;

  for (int t = 0; t < TT; ++t) {
    const unsigned short* xrow = xrowBase + (size_t)t * II;
    if (t + 1 < TT) __builtin_prefetch(xrow + II, 0, 0);   // next-step x row

    v8f acc = {};
#pragma unroll 4
    for (int k0 = 0; k0 < II; k0 += 32) {
      v16bf a = load_a_frag(xrow, k0, lane);
      v16bf b = load_b_frag(wihRow, k0, lane);
      acc = __builtin_amdgcn_wmma_f32_16x16x32_bf16(false, a, false, b,
                                                    (short)0, acc, false, false);
    }
    const unsigned short* hrow = hb + (size_t)rowB * HH;
#pragma unroll 4
    for (int k0 = 0; k0 < HH; k0 += 32) {
      v16bf a = load_a_frag(hrow, k0, lane);
      v16bf b = load_b_frag(whhRow, k0, lane);
      acc = __builtin_amdgcn_wmma_f32_16x16x32_bf16(false, a, false, b,
                                                    (short)0, acc, false, false);
    }
    // C/D layout: VGPR j -> row m0 + j + (lane>=16 ? 8 : 0), col n0 + lane%16.
    {
      const int col = n0 + l15;
      const int r0  = m0 + ((lane & 16) ? 8 : 0);
#pragma unroll
      for (int j = 0; j < 8; ++j) gates[(size_t)(r0 + j) * GG + col] = acc[j];
    }

    grid_barrier(bar, &gen);

    // Elementwise LSTM cell (bias folded here; c in registers).
#pragma unroll
    for (int e = 0; e < 2; ++e) {
      const int idx = gtid + e * 8192;
      const int b = idx >> 9;
      const int h = idx & (HH - 1);
      const float* gr = gates + (size_t)b * GG;
      float gi = gr[h]           + bih[h]           + bhh[h];
      float gf = gr[HH + h]      + bih[HH + h]      + bhh[HH + h];
      float gg = gr[2 * HH + h]  + bih[2 * HH + h]  + bhh[2 * HH + h];
      float go = gr[3 * HH + h]  + bih[3 * HH + h]  + bhh[3 * HH + h];
      gi = sigmoidf(gi);
      gf = sigmoidf(gf);
      gg = tanhf(gg);
      go = sigmoidf(go);
      float c = gf * creg[e] + gi * gg;
      creg[e] = c;
      float hn = go * tanhf(c);
      hb[idx] = f2bf(hn);
      if (t == TT - 1) out[(size_t)b * (2 * HH) + h] = fmaxf(hn, 0.0f);
    }

    grid_barrier(bar, &gen);
  }
}

// ---------------------------------------------------------------------------
// Kernel 3: backward direction = single LSTM step on x[:, T-1, :] from
// (h0[1], c0[1]). Plain f32 (~0.17 GFLOP). One thread per (b, h).
// ---------------------------------------------------------------------------
__global__ void bwd_one_step(const float* __restrict__ x,
                             const float* __restrict__ h0,
                             const float* __restrict__ c0,
                             const float* __restrict__ wihb,
                             const float* __restrict__ whhb,
                             const float* __restrict__ bihb,
                             const float* __restrict__ bhhb,
                             float* __restrict__ out) {
  const int idx = blockIdx.x * blockDim.x + threadIdx.x;
  if (idx >= BB * HH) return;
  const int b = idx >> 9;
  const int h = idx & (HH - 1);
  const float* xr = x + ((size_t)b * TT + (TT - 1)) * II;
  const float* hr = h0 + (size_t)BB * HH + (size_t)b * HH;  // dir 1
  const float* wi = wihb + (size_t)h * II;
  const float* wf = wihb + (size_t)(HH + h) * II;
  const float* wg = wihb + (size_t)(2 * HH + h) * II;
  const float* wo = wihb + (size_t)(3 * HH + h) * II;
  float gi = bihb[h]           + bhhb[h];
  float gf = bihb[HH + h]      + bhhb[HH + h];
  float gg = bihb[2 * HH + h]  + bhhb[2 * HH + h];
  float go = bihb[3 * HH + h]  + bhhb[3 * HH + h];
  for (int k = 0; k < II; ++k) {
    const float xv = xr[k];
    gi = fmaf(xv, wi[k], gi);
    gf = fmaf(xv, wf[k], gf);
    gg = fmaf(xv, wg[k], gg);
    go = fmaf(xv, wo[k], go);
  }
  const float* ui = whhb + (size_t)h * HH;
  const float* uf = whhb + (size_t)(HH + h) * HH;
  const float* ug = whhb + (size_t)(2 * HH + h) * HH;
  const float* uo = whhb + (size_t)(3 * HH + h) * HH;
  for (int k = 0; k < HH; ++k) {
    const float hv = hr[k];
    gi = fmaf(hv, ui[k], gi);
    gf = fmaf(hv, uf[k], gf);
    gg = fmaf(hv, ug[k], gg);
    go = fmaf(hv, uo[k], go);
  }
  gi = sigmoidf(gi);
  gf = sigmoidf(gf);
  gg = tanhf(gg);
  go = sigmoidf(go);
  const float c = gf * c0[(size_t)BB * HH + idx] + gi * gg;
  const float hn = go * tanhf(c);
  out[(size_t)b * (2 * HH) + HH + h] = fmaxf(hn, 0.0f);
}

// ---------------------------------------------------------------------------
// Launch. Workspace layout (~30.7 MB):
//   [0,256)        barrier counter
//   +262144        gates f32 [32][2048]
//   +32768         h buffer bf16 [32][512]
//   +25165824      x bf16 [32][512][768]
//   +3145728       W_ih_f bf16 [2048][768]
//   +2097152       W_hh_f bf16 [2048][512]
// ---------------------------------------------------------------------------
extern "C" void kernel_launch(void* const* d_in, const int* in_sizes, int n_in,
                              void* d_out, int out_size, void* d_ws, size_t ws_size,
                              hipStream_t stream) {
  const float* x    = (const float*)d_in[0];
  const float* h0   = (const float*)d_in[1];
  const float* c0   = (const float*)d_in[2];
  const float* wihf = (const float*)d_in[3];
  const float* whhf = (const float*)d_in[4];
  const float* bihf = (const float*)d_in[5];
  const float* bhhf = (const float*)d_in[6];
  const float* wihb = (const float*)d_in[7];
  const float* whhb = (const float*)d_in[8];
  const float* bihb = (const float*)d_in[9];
  const float* bhhb = (const float*)d_in[10];
  float* out = (float*)d_out;

  char* ws = (char*)d_ws;
  const size_t OFF_BAR   = 0;
  const size_t OFF_GATES = 256;
  const size_t OFF_HB    = OFF_GATES + (size_t)BB * GG * sizeof(float);
  const size_t OFF_XB    = OFF_HB + (size_t)BB * HH * sizeof(unsigned short);
  const size_t OFF_WIH   = OFF_XB + (size_t)BB * TT * II * sizeof(unsigned short);
  const size_t OFF_WHH   = OFF_WIH + (size_t)GG * II * sizeof(unsigned short);

  unsigned*       bar   = (unsigned*)(ws + OFF_BAR);
  float*          gates = (float*)(ws + OFF_GATES);
  unsigned short* hb    = (unsigned short*)(ws + OFF_HB);
  unsigned short* xb    = (unsigned short*)(ws + OFF_XB);
  unsigned short* wih   = (unsigned short*)(ws + OFF_WIH);
  unsigned short* whh   = (unsigned short*)(ws + OFF_WHH);

  const long long totalPrep = (long long)BB * TT * II + (long long)GG * II +
                              (long long)GG * HH + (long long)BB * HH;
  const int prepBlocks = (int)((totalPrep + 255) / 256);

  prep_kernel<<<prepBlocks, 256, 0, stream>>>(x, h0, wihf, whhf, xb, wih, whh, hb, bar);
  lstm_fwd_persistent<<<32, 256, 0, stream>>>(xb, wih, whh, hb, gates, bar,
                                              c0, bihf, bhhf, out);
  bwd_one_step<<<64, 256, 0, stream>>>(x, h0, c0, wihb, whhb, bihb, bhhb, out);
}
